// WinGNN_60928406061390
// MI455X (gfx1250) — matrix operand order repaired and checked by reading, not compiled, not run on hardware
//
#include <hip/hip_runtime.h>

// ---------------------------------------------------------------------------
// Types for CDNA5 WMMA (wave32, 16x16x32 bf16 -> f32)
// ---------------------------------------------------------------------------
typedef __attribute__((ext_vector_type(16))) __bf16 v16bf;
typedef __attribute__((ext_vector_type(8)))  float  v8f;
typedef __attribute__((ext_vector_type(4)))  int    v4i;
typedef __attribute__((address_space(1))) v4i v4i_as1;   // global
typedef __attribute__((address_space(3))) v4i v4i_as3;   // LDS

union Frag16 {
    v16bf        v;
    unsigned int u[8];
    uint4        q[2];
};

__device__ __forceinline__ v8f wmma_bf16(v16bf a, v16bf b, v8f c) {
    return __builtin_amdgcn_wmma_f32_16x16x32_bf16(
        /*neg_a=*/false, a, /*neg_b=*/false, b,
        /*c_mod=*/(short)0, c, /*reuse_a=*/false, /*reuse_b=*/false);
}

__device__ __forceinline__ unsigned short f2bf(float f) {
    unsigned int u = __float_as_uint(f);
    unsigned int r = (u + 0x7FFFu + ((u >> 16) & 1u)) >> 16;  // RNE
    return (unsigned short)r;
}
__device__ __forceinline__ unsigned int pk2(float a, float b) {
    return (unsigned int)f2bf(a) | ((unsigned int)f2bf(b) << 16);
}
__device__ __forceinline__ float bf2f(unsigned int lo16) {
    return __uint_as_float((lo16 & 0xFFFFu) << 16);
}

// ---------------------------------------------------------------------------
// CDNA5 async global->LDS copy (ASYNCcnt path), with portable fallback.
// ---------------------------------------------------------------------------
#if __has_builtin(__builtin_amdgcn_global_load_async_to_lds_b128) && \
    __has_builtin(__builtin_amdgcn_s_wait_asynccnt)
#define USE_ASYNC 1
#else
#define USE_ASYNC 0
#endif

__device__ __forceinline__ void cp16_g2l(unsigned short* lds_dst,
                                         const unsigned short* g_src) {
#if USE_ASYNC
    __builtin_amdgcn_global_load_async_to_lds_b128(
        (v4i_as1*)g_src, (v4i_as3*)lds_dst, 0, 0);
#else
    *(uint4*)lds_dst = *(const uint4*)g_src;
#endif
}
__device__ __forceinline__ void cp_wait(int pending_batches) {
#if USE_ASYNC
    if (pending_batches > 0) __builtin_amdgcn_s_wait_asynccnt(2);
    else                     __builtin_amdgcn_s_wait_asynccnt(0);
#else
    (void)pending_batches;
#endif
}

// ---------------------------------------------------------------------------
// Problem constants
// ---------------------------------------------------------------------------
#define NN     2048
#define DD     512
#define KNBR   5

// conv staging: padded bf16 image [3][70][72], K' = (ci,ky,kx) padded to 8 -> 192
#define IMG_PITCH 72
#define IMG_ROWS  70
#define SIMG_SZ   (3 * IMG_ROWS * IMG_PITCH)
#define KP        192
#define BW_SZ     (64 * KP)

__device__ __forceinline__ int grp_off(int g) {
    return (g < 21) ? ((g / 7) * 10080 + (g % 7) * 144) : 0;
}

// ---------------------------------------------------------------------------
// Kernel 1: conv7x7 s2 SAME + bias + relu + global-avg-pool  -> e[N][64] (f32)
// ---------------------------------------------------------------------------
__global__ __launch_bounds__(256) void conv_pool_kernel(
    const float* __restrict__ imgs, const float* __restrict__ conv_w,
    const float* __restrict__ conv_b, float* __restrict__ e)
{
    __shared__ __align__(16) unsigned short simg[SIMG_SZ];
    __shared__ __align__(16) unsigned short sw[BW_SZ];
    __shared__ float psum[8][64];

    const int tid = threadIdx.x;
    const int img = blockIdx.x;

    for (int i = tid; i < SIMG_SZ / 2; i += 256) ((unsigned int*)simg)[i] = 0u;
    for (int i = tid; i < BW_SZ  / 2; i += 256) ((unsigned int*)sw)[i]   = 0u;

    const float2* img2 = (const float2*)(imgs + (size_t)img * (3 * 64 * 64));
    for (int i = tid; i < 3 * 64 * 32; i += 256) {
        float2 v  = img2[i];
        int lin   = i * 2;
        int ci    = lin >> 12;
        int rem   = lin & 4095;
        int iy    = rem >> 6;
        int ix    = rem & 63;
        int o16   = (ci * IMG_ROWS + iy + 2) * IMG_PITCH + (ix + 2);
        *(unsigned int*)&simg[o16] = pk2(v.x, v.y);
    }
    for (int i = tid; i < 64 * 147; i += 256) {
        int n  = i / 147;
        int r  = i - n * 147;
        int ci = r / 49;
        int r2 = r - ci * 49;
        int ky = r2 / 7;
        int kx = r2 - ky * 7;
        sw[n * KP + ci * 56 + ky * 8 + kx] = f2bf(conv_w[i]);
    }
    __syncthreads();

    const int wave = tid >> 5;
    const int lane = tid & 31;
    const int half = lane >> 4;
    const int l15  = lane & 15;

    float bias[4];
#pragma unroll
    for (int nt = 0; nt < 4; ++nt) bias[nt] = conv_b[nt * 16 + l15];

    float colsum[4] = {0.f, 0.f, 0.f, 0.f};

    for (int mt = wave; mt < 64; mt += 8) {
        const int m     = mt * 16 + l15;
        const int oy    = m >> 5;
        const int ox    = m & 31;
        const int vbase = oy * 288 + ox * 4;

        v8f acc[4] = {};
#pragma unroll
        for (int s = 0; s < 6; ++s) {
            Frag16 a;
            const int off0 = vbase + (half ? grp_off(4 * s + 1) : grp_off(4 * s));
            const int off1 = vbase + (half ? grp_off(4 * s + 3) : grp_off(4 * s + 2));
#pragma unroll
            for (int j = 0; j < 4; ++j) {
                a.u[j]     = *(const unsigned int*)&simg[(off0 >> 1) + j * 2];
                a.u[4 + j] = *(const unsigned int*)&simg[(off1 >> 1) + j * 2];
            }
#pragma unroll
            for (int nt = 0; nt < 4; ++nt) {
                Frag16 b;
                const int c16 = (nt * 16 + l15) * KP + s * 32 + half * 8;
                b.q[0] = *(const uint4*)&sw[c16];
                b.q[1] = *(const uint4*)&sw[c16 + 16];
                acc[nt] = wmma_bf16(a.v, b.v, acc[nt]);
            }
        }
#pragma unroll
        for (int nt = 0; nt < 4; ++nt)
#pragma unroll
            for (int r = 0; r < 8; ++r)
                colsum[nt] += fmaxf(acc[nt][r] + bias[nt], 0.f);
    }

#pragma unroll
    for (int nt = 0; nt < 4; ++nt) {
        float v = colsum[nt] + __shfl_down(colsum[nt], 16, 32);
        if (half == 0) psum[wave][nt * 16 + l15] = v;
    }
    __syncthreads();
    if (tid < 64) {
        float s = 0.f;
#pragma unroll
        for (int w = 0; w < 8; ++w) s += psum[w][tid];
        e[(size_t)img * 64 + tid] = s * (1.f / 1024.f);
    }
}

// ---------------------------------------------------------------------------
// Kernel 2: projection  h_bf16 = relu(e[N][64] @ proj_w[64][512] + proj_b)
// ---------------------------------------------------------------------------
__global__ __launch_bounds__(256) void proj_kernel(
    const float* __restrict__ e, const float* __restrict__ pw,
    const float* __restrict__ pb, unsigned short* __restrict__ hbf)
{
    __shared__ __align__(16) unsigned short sa[16 * 64];
    __shared__ __align__(16) unsigned short sb[128 * 64];

    const int tid  = threadIdx.x;
    const int wave = tid >> 5, lane = tid & 31;
    const int half = lane >> 4, l15 = lane & 15;
    const int m0 = blockIdx.x * 16;
    const int n0 = blockIdx.y * 128;

    {
        int r  = tid >> 4;
        int k4 = (tid & 15) * 4;
        float4 v = *(const float4*)&e[(size_t)(m0 + r) * 64 + k4];
        *(unsigned int*)&sa[r * 64 + k4]     = pk2(v.x, v.y);
        *(unsigned int*)&sa[r * 64 + k4 + 2] = pk2(v.z, v.w);
    }
    for (int i = tid; i < 128 * 64; i += 256) {
        int col = i & 127, k = i >> 7;
        sb[col * 64 + k] = f2bf(pw[(size_t)k * 512 + n0 + col]);
    }
    __syncthreads();

    v8f acc = {};
#pragma unroll
    for (int s = 0; s < 2; ++s) {
        Frag16 a, b;
        const int abase = l15 * 64 + s * 32 + half * 8;
        a.q[0] = *(const uint4*)&sa[abase];
        a.q[1] = *(const uint4*)&sa[abase + 16];
        const int bbase = (wave * 16 + l15) * 64 + s * 32 + half * 8;
        b.q[0] = *(const uint4*)&sb[bbase];
        b.q[1] = *(const uint4*)&sb[bbase + 16];
        acc = wmma_bf16(a.v, b.v, acc);
    }
    const int c  = n0 + wave * 16 + l15;
    const float bv = pb[c];
#pragma unroll
    for (int r = 0; r < 8; ++r) {
        int m = m0 + r + half * 8;
        hbf[(size_t)m * 512 + c] = f2bf(fmaxf(acc[r] + bv, 0.f));
    }
}

// ---------------------------------------------------------------------------
// Kernel 3: pre-transpose + convert GNN weights:
//   wT[l][mat][n][k] (bf16) = w_{self,nbr}[l][k][n] (f32)
// ---------------------------------------------------------------------------
__global__ __launch_bounds__(256) void cvt_wT_kernel(
    const float* __restrict__ w_self, const float* __restrict__ w_nbr,
    unsigned short* __restrict__ wT)
{
    const int b   = blockIdx.x;          // ((l*2 + mat)*512 + n)
    const int l   = b >> 10;
    const int mat = (b >> 9) & 1;
    const int n   = b & 511;
    const float* src = (mat ? w_nbr : w_self) + (size_t)l * DD * DD;
    unsigned short* dst = wT + (size_t)b * DD;
    const int k = threadIdx.x * 2;
    *(unsigned int*)&dst[k] = pk2(src[(size_t)k * DD + n], src[(size_t)(k + 1) * DD + n]);
}

// ---------------------------------------------------------------------------
// Kernel 4: radius-KNN (r^2 = 2, K = 5, no self loops)
// ---------------------------------------------------------------------------
__global__ __launch_bounds__(256) void knn_kernel(
    const float* __restrict__ pos, int* __restrict__ idx, float* __restrict__ wrec)
{
    __shared__ float2 sp[NN];
    const int tid = threadIdx.x;
    const float2* p2 = (const float2*)pos;
    for (int t = tid; t < NN; t += 256) {
        float2 p = p2[t];
        sp[t] = make_float2(p.x * 32.0f, p.y * 32.0f);
    }
    __syncthreads();

    const int i = blockIdx.x * 256 + tid;
    const float2 pi = sp[i];
    float bd[KNBR];
    int   bi[KNBR];
#pragma unroll
    for (int k = 0; k < KNBR; ++k) { bd[k] = 1e30f; bi[k] = -1; }

    for (int j = 0; j < NN; ++j) {
        float dx = pi.x - sp[j].x;
        float dy = pi.y - sp[j].y;
        float d2 = dx * dx + dy * dy;
        if (j != i && d2 <= 2.0f && d2 < bd[KNBR - 1]) {
            float cd = d2; int cj = j;
#pragma unroll
            for (int k = 0; k < KNBR; ++k) {
                if (cd < bd[k]) {
                    float td = bd[k]; int tj = bi[k];
                    bd[k] = cd; bi[k] = cj; cd = td; cj = tj;
                }
            }
        }
    }
    int cnt = 0;
#pragma unroll
    for (int k = 0; k < KNBR; ++k) {
        idx[i * KNBR + k] = bi[k];
        cnt += (bi[k] >= 0) ? 1 : 0;
    }
    wrec[i] = 1.0f / (float)max(cnt, 1);
}

// ---------------------------------------------------------------------------
// Kernel 5: neighbor mean-aggregate (bf16 in, bf16 out)
// ---------------------------------------------------------------------------
__global__ __launch_bounds__(128) void agg_kernel(
    const unsigned short* __restrict__ h, const int* __restrict__ idx,
    const float* __restrict__ wrec, unsigned short* __restrict__ agg)
{
    const int i = blockIdx.x;
    const int c = threadIdx.x * 4;
    float s0 = 0.f, s1 = 0.f, s2 = 0.f, s3 = 0.f;
#pragma unroll
    for (int k = 0; k < KNBR; ++k) {
        int j = idx[i * KNBR + k];
        if (j >= 0) {
            uint2 v = *(const uint2*)&h[(size_t)j * 512 + c];
            s0 += bf2f(v.x); s1 += bf2f(v.x >> 16);
            s2 += bf2f(v.y); s3 += bf2f(v.y >> 16);
        }
    }
    float w = wrec[i];
    uint2 o;
    o.x = pk2(s0 * w, s1 * w);
    o.y = pk2(s2 * w, s3 * w);
    *(uint2*)&agg[(size_t)i * 512 + c] = o;
}

// ---------------------------------------------------------------------------
// Kernel 6: GNN layer GEMM, fused K = 1024 ( [h|agg] x [WselfT;WnbrT] )
// Block = 64M x 64N, 8 waves (2 mi x 4 nj), each wave 2 M-tiles.
// Double-buffered raw-bf16 tile staging via async global->LDS (b128).
// ---------------------------------------------------------------------------
__global__ __launch_bounds__(256) void gnn_gemm_kernel(
    const unsigned short* __restrict__ h, const unsigned short* __restrict__ agg,
    const unsigned short* __restrict__ wT,   // [2][512 n][512 k] bf16
    const float* __restrict__ bias,
    float* __restrict__ outf, unsigned short* __restrict__ outbf, int write_f32)
{
    __shared__ __align__(16) unsigned short sa[2][64 * 32];
    __shared__ __align__(16) unsigned short sb[2][64 * 32];

    const int tid  = threadIdx.x;
    const int wave = tid >> 5, lane = tid & 31;
    const int half = lane >> 4, l15 = lane & 15;
    const int mi = wave & 1, nj = wave >> 1;
    const int m0 = blockIdx.x * 64;
    const int n0 = blockIdx.y * 64;

    const int crow = tid >> 2;        // 0..63 (A row / B col)
    const int cseg = (tid & 3) * 8;   // 16B segment within the 64B row

    auto issue = [&](int t) {
        const int buf = t & 1;
        const int ph  = t >> 4;
        const int k0  = (t & 15) * 32;
        const unsigned short* srcA = ph ? agg : h;
        const unsigned short* srcB = wT + (size_t)ph * (DD * DD);
        cp16_g2l(&sa[buf][crow * 32 + cseg],
                 &srcA[(size_t)(m0 + crow) * 512 + k0 + cseg]);
        cp16_g2l(&sb[buf][crow * 32 + cseg],
                 &srcB[(size_t)(n0 + crow) * 512 + k0 + cseg]);
    };

    v8f acc[2] = {};
    issue(0);
    for (int t = 0; t < 32; ++t) {
        const int buf = t & 1;
        if (t + 1 < 32) issue(t + 1);
        cp_wait((t + 1 < 32) ? 1 : 0);
        __syncthreads();              // tile t ready in LDS for all waves

        Frag16 b;
        const int bbase = (nj * 16 + l15) * 32 + half * 8;
        b.q[0] = *(const uint4*)&sb[buf][bbase];
        b.q[1] = *(const uint4*)&sb[buf][bbase + 16];
#pragma unroll
        for (int mt = 0; mt < 2; ++mt) {
            Frag16 a;
            const int abase = (mi * 32 + mt * 16 + l15) * 32 + half * 8;
            a.q[0] = *(const uint4*)&sa[buf][abase];
            a.q[1] = *(const uint4*)&sa[buf][abase + 16];
            acc[mt] = wmma_bf16(a.v, b.v, acc[mt]);
        }
        __syncthreads();              // everyone done reading before overwrite
    }

    const int c  = n0 + nj * 16 + l15;
    const float bv = bias[c];
#pragma unroll
    for (int mt = 0; mt < 2; ++mt) {
#pragma unroll
        for (int r = 0; r < 8; ++r) {
            int m = m0 + mi * 32 + mt * 16 + r + half * 8;
            float v = fmaxf(acc[mt][r] + bv, 0.f);
            if (write_f32) outf[(size_t)m * 512 + c] = v;
            else           outbf[(size_t)m * 512 + c] = f2bf(v);
        }
    }
}

// ---------------------------------------------------------------------------
// Host launcher
// ---------------------------------------------------------------------------
extern "C" void kernel_launch(void* const* d_in, const int* in_sizes, int n_in,
                              void* d_out, int out_size, void* d_ws, size_t ws_size,
                              hipStream_t stream)
{
    (void)in_sizes; (void)n_in; (void)out_size; (void)ws_size;
    const float* imgs   = (const float*)d_in[0];
    const float* pos    = (const float*)d_in[1];
    const float* conv_w = (const float*)d_in[2];
    const float* conv_b = (const float*)d_in[3];
    const float* proj_w = (const float*)d_in[4];
    const float* proj_b = (const float*)d_in[5];
    const float* w_self = (const float*)d_in[6];
    const float* w_nbr  = (const float*)d_in[7];
    const float* b_gnn  = (const float*)d_in[8];
    float* out = (float*)d_out;

    char* ws = (char*)d_ws;
    auto carve = [&](size_t bytes) {
        char* p = ws;
        ws += (bytes + 255) & ~(size_t)255;
        return p;
    };
    float*          e    = (float*)         carve((size_t)NN * 64 * 4);
    unsigned short* hA   = (unsigned short*)carve((size_t)NN * DD * 2);
    unsigned short* hB   = (unsigned short*)carve((size_t)NN * DD * 2);
    unsigned short* aggb = (unsigned short*)carve((size_t)NN * DD * 2);
    unsigned short* wT   = (unsigned short*)carve((size_t)4 * 2 * DD * DD * 2);
    int*            idx  = (int*)           carve((size_t)NN * KNBR * 4);
    float*          wrec = (float*)         carve((size_t)NN * 4);

    conv_pool_kernel<<<NN, 256, 0, stream>>>(imgs, conv_w, conv_b, e);
    proj_kernel<<<dim3(NN / 16, DD / 128), 256, 0, stream>>>(e, proj_w, proj_b, hA);
    cvt_wT_kernel<<<4 * 2 * DD, 256, 0, stream>>>(w_self, w_nbr, wT);
    knn_kernel<<<NN / 256, 256, 0, stream>>>(pos, idx, wrec);

    unsigned short* cur = hA;
    unsigned short* nxt = hB;
    for (int l = 0; l < 4; ++l) {
        agg_kernel<<<NN, 128, 0, stream>>>(cur, idx, wrec, aggb);
        int last = (l == 3);
        gnn_gemm_kernel<<<dim3(NN / 64, DD / 64), 256, 0, stream>>>(
            cur, aggb, wT + (size_t)l * 2 * DD * DD,
            b_gnn + (size_t)l * DD,
            last ? out : nullptr, last ? nullptr : nxt, last);
        unsigned short* t = cur; cur = nxt; nxt = t;
    }
}
